// MultiHeadAttention_2345052144119
// MI455X (gfx1250) — compile-verified, matrix operand order checked by reading, not run on hardware
//
#include <hip/hip_runtime.h>
#include <hip/hip_bf16.h>

// ---------------------------------------------------------------------------
// Multi-head attention, CDNA5 (gfx1250) WMMA implementation.
//   B=8, S=2048, E=640, H=5, DH=128
// Pipeline:
//   0) cvt kernels     : x, w_qkv, w_proj -> bf16 once
//   1) qkv_gemm_kernel : xb @ wb^T + b_qkv -> Q,K (row-major bf16),
//                        V stored transposed [DH,S] bf16.  32x64 tile/wave.
//   2) attn_kernel     : flash-style attention; 4 waves/block; K/V tiles
//                        double-buffered in LDS via global_load_async_to_lds,
//                        matmuls via v_wmma_f32_16x16x32_bf16 (frags from LDS)
//   3) proj_gemm_kernel: ctx(bf16) @ w_proj^T + b_proj -> out (f32)
// ---------------------------------------------------------------------------

#define B_   8
#define S_   2048
#define E_   640
#define H_   5
#define DH_  128
#define E3_  1920

typedef __attribute__((ext_vector_type(16))) __bf16 v16bf;
typedef __attribute__((ext_vector_type(8)))  __bf16 v8bf;
typedef __attribute__((ext_vector_type(8)))  float  v8f;

// 16x32 (16-bit) WMMA A/B fragment for this lane.
// Lanes 0-15 hold K = {base+0..7, base+16..23}, lanes 16-31 hold
// K = {base+8..15, base+24..31}; caller folds (lane>>4)*8 into the pointer.
__device__ inline v16bf frag_from_bf16(const __bf16* p) {
  union { v16bf v; v8bf h[2]; } u;
  u.h[0] = *(const v8bf*)(p);
  u.h[1] = *(const v8bf*)(p + 16);
  return u.v;
}

#define WMMA_BF16(A, Bm, C) \
  __builtin_amdgcn_wmma_f32_16x16x32_bf16(false, (A), false, (Bm), (short)0, (C), false, false)

// Async copy: 16 bytes global -> LDS per lane (CDNA5 async path, ASYNCcnt).
__device__ inline void async_load_b128(const __bf16* gaddr, __bf16* laddr) {
  const unsigned lds_off = (unsigned)(uintptr_t)laddr;   // addr[31:0] = LDS byte offset
  asm volatile("global_load_async_to_lds_b128 %0, %1, off"
               :: "v"(lds_off), "v"((unsigned long long)(uintptr_t)gaddr)
               : "memory");
}
__device__ inline void wait_async0() {
  asm volatile("s_wait_asynccnt 0" ::: "memory");
}

// ---------------------------------------------------------------------------
// Kernel 0: f32 -> bf16 conversion (sizes are multiples of 8)
// ---------------------------------------------------------------------------
__global__ __launch_bounds__(256)
void cvt_f32_bf16_kernel(const float* __restrict__ in, __bf16* __restrict__ out, int n) {
  const int i = (blockIdx.x * 256 + threadIdx.x) * 8;
  if (i < n) {
    const float4 a = *(const float4*)(in + i);
    const float4 b = *(const float4*)(in + i + 4);
    union { v8bf v; __bf16 e[8]; } u;
    u.e[0] = (__bf16)a.x; u.e[1] = (__bf16)a.y; u.e[2] = (__bf16)a.z; u.e[3] = (__bf16)a.w;
    u.e[4] = (__bf16)b.x; u.e[5] = (__bf16)b.y; u.e[6] = (__bf16)b.z; u.e[7] = (__bf16)b.w;
    *(v8bf*)(out + i) = u.v;
  }
}

// ---------------------------------------------------------------------------
// Kernel 1: QKV projection.  Grid (512, 30), 32 threads/block, 32x64 tile.
// ---------------------------------------------------------------------------
__global__ __launch_bounds__(32)
void qkv_gemm_kernel(const __bf16* __restrict__ xb,    // [B*S, E] bf16
                     const __bf16* __restrict__ wb,    // [3E, E]  bf16
                     const float*  __restrict__ bias,  // [3E]
                     __bf16* __restrict__ Qb,          // [B*H, S, DH]
                     __bf16* __restrict__ Kb,          // [B*H, S, DH]
                     __bf16* __restrict__ Vt)          // [B*H, DH, S]
{
  const int lane = threadIdx.x;
  const int lm = lane & 15;
  const int lh = lane >> 4;
  const int m0 = blockIdx.x * 32;
  const int n0 = blockIdx.y * 64;

  v8f acc[2][4];
#pragma unroll
  for (int mi = 0; mi < 2; ++mi)
#pragma unroll
    for (int t = 0; t < 4; ++t)
#pragma unroll
      for (int r = 0; r < 8; ++r) acc[mi][t][r] = 0.0f;

  const __bf16* arow0 = xb + (size_t)(m0 + lm) * E_ + lh * 8;
  const __bf16* arow1 = arow0 + (size_t)16 * E_;

  for (int kc = 0; kc < E_; kc += 32) {
    const v16bf a0 = frag_from_bf16(arow0 + kc);
    const v16bf a1 = frag_from_bf16(arow1 + kc);
#pragma unroll
    for (int t = 0; t < 4; ++t) {
      const v16bf bm = frag_from_bf16(wb + (size_t)(n0 + t * 16 + lm) * E_ + kc + lh * 8);
      acc[0][t] = WMMA_BF16(a0, bm, acc[0][t]);
      acc[1][t] = WMMA_BF16(a1, bm, acc[1][t]);
    }
  }

#pragma unroll
  for (int mi = 0; mi < 2; ++mi) {
#pragma unroll
    for (int t = 0; t < 4; ++t) {
      const int n = n0 + t * 16 + lm;          // feature in [0, 3E)
      const float bb = bias[n];
#pragma unroll
      for (int r = 0; r < 8; ++r) {
        const int token = m0 + mi * 16 + r + 8 * lh;
        const int bi = token >> 11;            // / S
        const int s  = token & (S_ - 1);
        const __bf16 bv = (__bf16)(acc[mi][t][r] + bb);
        if (n < E_) {
          const int hh = n / DH_, d = n & (DH_ - 1);
          Qb[(((size_t)bi * H_ + hh) * S_ + s) * DH_ + d] = bv;
        } else if (n < 2 * E_) {
          const int e = n - E_;
          const int hh = e / DH_, d = e & (DH_ - 1);
          Kb[(((size_t)bi * H_ + hh) * S_ + s) * DH_ + d] = bv;
        } else {
          const int e = n - 2 * E_;
          const int hh = e / DH_, d = e & (DH_ - 1);
          Vt[(((size_t)bi * H_ + hh) * DH_ + d) * S_ + s] = bv;   // transposed
        }
      }
    }
  }
}

// ---------------------------------------------------------------------------
// Kernel 2: flash-style attention.
// Grid = B*H*(S/64) = 1280 blocks, 128 threads (4 waves).  Each block owns 64
// query rows of one (b,h); each wave owns 16.  K/V tiles are double-buffered
// in LDS via async global->LDS copies overlapping the WMMA stream.
// ---------------------------------------------------------------------------
#define QW 4                      // waves per block
#define KP (DH_ + 8)              // padded K-tile row (bank-conflict free)
#define VP (32 + 8)               // padded V-tile row
#define PP (32 + 8)               // padded P row

__global__ __launch_bounds__(128)
void attn_kernel(const __bf16* __restrict__ Qb,
                 const __bf16* __restrict__ Kb,
                 const __bf16* __restrict__ Vt,
                 __bf16* __restrict__ Ctx)   // [B, S, E] bf16
{
  __shared__ __bf16 sK[2][32][KP];     // 2 x (32 keys x 128 dh)  (~17.4 KB)
  __shared__ __bf16 sV[2][DH_][VP];    // 2 x (128 dh  x 32 keys) (~20.5 KB)
  __shared__ float  sS[QW][16][33];    // per-wave raw scores     (~8.4 KB)
  __shared__ __bf16 sP[QW][16][PP];    // per-wave exp(scores)    (~5.1 KB)
  __shared__ float  rowm[QW][16], rowl[QW][16], rfac[QW][16];

  const int tid  = threadIdx.x;
  const int w    = tid >> 5;
  const int lane = tid & 31;
  const int lm   = lane & 15;
  const int lh   = lane >> 4;

  const int qtiles64 = S_ / 64;
  const int bh  = blockIdx.x / qtiles64;
  const int q64 = blockIdx.x % qtiles64;
  const int bi  = bh / H_;
  const int h   = bh % H_;

  const __bf16* Qp = Qb + (size_t)bh * S_ * DH_;
  const __bf16* Kp = Kb + (size_t)bh * S_ * DH_;
  const __bf16* Vp = Vt + (size_t)bh * DH_ * S_;

  const int qrow0 = q64 * 64 + w * 16;

  // Q fragments (16 rows x 128 -> 4 fragments), resident in VGPRs.
  v16bf qf[4];
  const __bf16* qrow = Qp + (size_t)(qrow0 + lm) * DH_ + lh * 8;
#pragma unroll
  for (int c = 0; c < 4; ++c) qf[c] = frag_from_bf16(qrow + c * 32);

  v8f oacc[8];
#pragma unroll
  for (int t = 0; t < 8; ++t)
#pragma unroll
    for (int r = 0; r < 8; ++r) oacc[t][r] = 0.0f;

  if (lane < 16) { rowm[w][lane] = -1e30f; rowl[w][lane] = 0.0f; }  // wave-private

  // Stage K tile (32x128) + V^T tile (128x32) asynchronously: 4+4 chunks of
  // 16B per thread (512 chunks each across 128 threads).
  auto stage = [&](int kt, int p) {
#pragma unroll
    for (int it = 0; it < 4; ++it) {
      const int chunk = tid + it * 128;            // 0..511
      const int kr  = chunk >> 4;                  // key row 0..31
      const int kcc = chunk & 15;                  // dh chunk
      async_load_b128(Kp + (size_t)(kt + kr) * DH_ + kcc * 8, &sK[p][kr][kcc * 8]);
      const int vr  = chunk >> 2;                  // dh row 0..127
      const int vcc = chunk & 3;                   // key chunk
      async_load_b128(Vp + (size_t)vr * S_ + kt + vcc * 8, &sV[p][vr][vcc * 8]);
    }
  };

  stage(0, 0);   // prologue: tile 0 in flight

  for (int kt = 0; kt < S_; kt += 32) {
    const int p = (kt >> 5) & 1;

    wait_async0();        // my async copies for tile kt have landed in LDS
    __syncthreads();      // everyone's copies landed; everyone done reading p^1

    if (kt + 32 < S_) stage(kt + 32, p ^ 1);   // overlap next tile's copy

    // ---- S_ij = Q (16x128) . K^T (128x32): fragments from LDS ------------
    v8f sacc[2];
#pragma unroll
    for (int t = 0; t < 2; ++t)
#pragma unroll
      for (int r = 0; r < 8; ++r) sacc[t][r] = 0.0f;

#pragma unroll
    for (int c = 0; c < 4; ++c) {
#pragma unroll
      for (int t = 0; t < 2; ++t) {
        const v16bf kf = frag_from_bf16(&sK[p][t * 16 + lm][c * 32 + lh * 8]);
        sacc[t] = WMMA_BF16(qf[c], kf, sacc[t]);
      }
    }

    // ---- spill scores (wave-private region; wave LDS ops are in-order) ---
#pragma unroll
    for (int t = 0; t < 2; ++t)
#pragma unroll
      for (int r = 0; r < 8; ++r)
        sS[w][r + 8 * lh][t * 16 + lm] = sacc[t][r];

    // ---- online softmax: lanes 0..15 each own one query row --------------
    if (lane < 16) {
      const float mold = rowm[w][lane];
      float mx = mold;
#pragma unroll
      for (int j = 0; j < 32; ++j) mx = fmaxf(mx, sS[w][lane][j]);
      const float fac = __expf(mold - mx);
      float lsum = rowl[w][lane] * fac;
#pragma unroll
      for (int j = 0; j < 32; ++j) {
        const float p2 = __expf(sS[w][lane][j] - mx);
        sP[w][lane][j] = (__bf16)p2;
        lsum += p2;
      }
      rowm[w][lane] = mx;
      rowl[w][lane] = lsum;
      rfac[w][lane] = fac;
    }

    // ---- rescale O accumulators, then O += P.V ---------------------------
    float fr[8];
#pragma unroll
    for (int r = 0; r < 8; ++r) fr[r] = rfac[w][r + 8 * lh];

    const v16bf pf = frag_from_bf16(&sP[w][lm][lh * 8]);   // 16x32 A-frag

#pragma unroll
    for (int t = 0; t < 8; ++t) {
#pragma unroll
      for (int r = 0; r < 8; ++r) oacc[t][r] *= fr[r];
      const v16bf vf = frag_from_bf16(&sV[p][t * 16 + lm][lh * 8]);
      oacc[t] = WMMA_BF16(pf, vf, oacc[t]);
    }
    // no trailing barrier: next iteration's staging targets the other buffer
  }

  // ---- epilogue: normalize by row sums, store context (bf16) --------------
  float rl[8];
#pragma unroll
  for (int r = 0; r < 8; ++r) rl[r] = rowl[w][r + 8 * lh];

#pragma unroll
  for (int t = 0; t < 8; ++t) {
#pragma unroll
    for (int r = 0; r < 8; ++r) {
      const int row = r + 8 * lh;
      const float val = oacc[t][r] / rl[r];
      Ctx[((size_t)bi * S_ + (qrow0 + row)) * E_ + h * DH_ + t * 16 + lm] = (__bf16)val;
    }
  }
}

// ---------------------------------------------------------------------------
// Kernel 3: output projection.  Grid (512, 10), 32 threads/block, 32x64 tile.
// ---------------------------------------------------------------------------
__global__ __launch_bounds__(32)
void proj_gemm_kernel(const __bf16* __restrict__ ctx,  // [B*S, E] bf16
                      const __bf16* __restrict__ wb,   // [E, E] bf16
                      const float*  __restrict__ bias, // [E]
                      float* __restrict__ out)         // [B*S, E] f32
{
  const int lane = threadIdx.x;
  const int lm = lane & 15;
  const int lh = lane >> 4;
  const int m0 = blockIdx.x * 32;
  const int n0 = blockIdx.y * 64;

  v8f acc[2][4];
#pragma unroll
  for (int mi = 0; mi < 2; ++mi)
#pragma unroll
    for (int t = 0; t < 4; ++t)
#pragma unroll
      for (int r = 0; r < 8; ++r) acc[mi][t][r] = 0.0f;

  const __bf16* arow0 = ctx + (size_t)(m0 + lm) * E_ + lh * 8;
  const __bf16* arow1 = arow0 + (size_t)16 * E_;

  for (int kc = 0; kc < E_; kc += 32) {
    const v16bf a0 = frag_from_bf16(arow0 + kc);
    const v16bf a1 = frag_from_bf16(arow1 + kc);
#pragma unroll
    for (int t = 0; t < 4; ++t) {
      const v16bf bm = frag_from_bf16(wb + (size_t)(n0 + t * 16 + lm) * E_ + kc + lh * 8);
      acc[0][t] = WMMA_BF16(a0, bm, acc[0][t]);
      acc[1][t] = WMMA_BF16(a1, bm, acc[1][t]);
    }
  }

#pragma unroll
  for (int mi = 0; mi < 2; ++mi) {
#pragma unroll
    for (int t = 0; t < 4; ++t) {
      const int n = n0 + t * 16 + lm;
      const float bb = bias[n];
#pragma unroll
      for (int r = 0; r < 8; ++r) {
        const int token = m0 + mi * 16 + r + 8 * lh;
        out[(size_t)token * E_ + n] = acc[mi][t][r] + bb;
      }
    }
  }
}

// ---------------------------------------------------------------------------
extern "C" void kernel_launch(void* const* d_in, const int* in_sizes, int n_in,
                              void* d_out, int out_size, void* d_ws, size_t ws_size,
                              hipStream_t stream) {
  const float* x      = (const float*)d_in[0];   // [8, 2048, 640]
  const float* w_qkv  = (const float*)d_in[1];   // [1920, 640]
  const float* b_qkv  = (const float*)d_in[2];   // [1920]
  const float* w_proj = (const float*)d_in[3];   // [640, 640]
  const float* b_proj = (const float*)d_in[4];   // [640]
  float* out = (float*)d_out;                    // [8, 2048, 640]

  // Workspace partition (all bf16), ~108 MB total.
  const size_t nTok = (size_t)B_ * S_ * E_;      // 10,485,760
  const size_t nQ   = (size_t)B_ * H_ * S_ * DH_;// 10,485,760
  const size_t nWq  = (size_t)E3_ * E_;          // 1,228,800
  const size_t nWp  = (size_t)E_ * E_;           // 409,600
  __bf16* Qb  = (__bf16*)d_ws;
  __bf16* Kb  = Qb  + nQ;
  __bf16* Vt  = Kb  + nQ;
  __bf16* Ctx = Vt  + nQ;
  __bf16* xb  = Ctx + nTok;
  __bf16* wqb = xb  + nTok;
  __bf16* wpb = wqb + nWq;

  // 0) one-shot f32 -> bf16 conversions
  cvt_f32_bf16_kernel<<<dim3((int)(nTok / 8 / 256)), dim3(256), 0, stream>>>(x, xb, (int)nTok);
  cvt_f32_bf16_kernel<<<dim3((int)(nWq  / 8 / 256)), dim3(256), 0, stream>>>(w_qkv, wqb, (int)nWq);
  cvt_f32_bf16_kernel<<<dim3((int)(nWp  / 8 / 256)), dim3(256), 0, stream>>>(w_proj, wpb, (int)nWp);

  dim3 wv(32, 1, 1);

  // 1) QKV projection: M = 16384, N = 1920
  qkv_gemm_kernel<<<dim3((B_ * S_) / 32, E3_ / 64, 1), wv, 0, stream>>>(
      xb, wqb, b_qkv, Qb, Kb, Vt);

  // 2) attention: 4-wave blocks, 64 query rows per block
  attn_kernel<<<dim3(B_ * H_ * (S_ / 64), 1, 1), dim3(128, 1, 1), 0, stream>>>(
      Qb, Kb, Vt, Ctx);

  // 3) output projection: M = 16384, N = 640
  proj_gemm_kernel<<<dim3((B_ * S_) / 32, E_ / 64, 1), wv, 0, stream>>>(
      Ctx, wpb, b_proj, out);
}
